// Wavenet_Student_5609227289040
// MI455X (gfx1250) — matrix operand order repaired and checked by reading, not compiled, not run on hardware
//
#include <hip/hip_runtime.h>

// ---------------------------------------------------------------------------
// Types
// ---------------------------------------------------------------------------
typedef _Float16 v16h __attribute__((ext_vector_type(16)));
typedef _Float16 v8h  __attribute__((ext_vector_type(8)));
typedef float    v8f  __attribute__((ext_vector_type(8)));
typedef unsigned int v4u __attribute__((ext_vector_type(4)));
typedef int      v8i  __attribute__((ext_vector_type(8)));
typedef int      v4i  __attribute__((ext_vector_type(4)));

#define DEV __device__ static inline

namespace cfg {
constexpr int B = 4, T = 16000;
constexpr int RC = 128, GC = 256, SC = 128, CIN = 80, CINP = 96;
constexpr int KFG = 3 * RC + CINP;            // 480 = gate GEMM K (3 taps + padded cond)
constexpr int NT  = T / 16;                   // 1000 time tiles of 16
constexpr float SQRT_HALF = 0.70710678118654752440f;
}

#if __has_builtin(__builtin_amdgcn_tensor_load_to_lds)
#define USE_TDM 1
#else
#define USE_TDM 0
#endif

// ---------------------------------------------------------------------------
// WMMA fragment helpers (A and B use contiguous k order k = 16*(L/16) + s, wave32)
// ---------------------------------------------------------------------------
DEV v8f zero8() { v8f z = {0.f,0.f,0.f,0.f,0.f,0.f,0.f,0.f}; return z; }

DEV v8f wmma_f16(v16h a, v16h b, v8f c) {
  return __builtin_amdgcn_wmma_f32_16x16x32_f16(false, a, false, b, (short)0, c,
                                                false, false);
}

// A-matrix 16x32 f16 fragment from row-major [M x lda] (lda in halves).
// Lane L: M = m_base + L%16. halves 0..15 = K[k0+16*(L>=16) .. +15]
// (contiguous, matching load_b_frag).
DEV v16h load_a_frag(const _Float16* A, int lda, int m, int k0, int lane) {
  const int hi = lane >> 4;
  const _Float16* p = A + (size_t)m * lda + k0 + hi * 16;
  v8h lo = *(const v8h*)p;
  v8h hh = *(const v8h*)(p + 8);
  v16h r;
#pragma unroll
  for (int i = 0; i < 8; ++i) { r[i] = lo[i]; r[8 + i] = hh[i]; }
  return r;
}

// B-matrix 32x16 f16 fragment from LDS stored as [n][k] with row stride `str`
// halves. Lane L: N = L%16, K run = k0 + 16*(L>=16) .. +15 (contiguous).
DEV v16h load_b_frag(const _Float16* bb, int str, int k0, int lane) {
  const int n = lane & 15, hi = lane >> 4;
  const _Float16* p = bb + n * str + k0 + hi * 16;
  v8h lo = *(const v8h*)p;
  v8h hh = *(const v8h*)(p + 8);
  v16h r;
#pragma unroll
  for (int i = 0; i < 8; ++i) { r[i] = lo[i]; r[8 + i] = hh[i]; }
  return r;
}

// ---------------------------------------------------------------------------
// Weight / input packing kernels (run once per launch; all f16 conversion)
// ---------------------------------------------------------------------------
__global__ void pack_c_kernel(const float* __restrict__ c, _Float16* __restrict__ c16) {
  using namespace cfg;
  int idx = blockIdx.x * 256 + threadIdx.x;
  int tot = B * CINP * T;
  if (idx >= tot) return;
  int t = idx % T;
  int r = idx / T;
  int cin = r % CINP;
  int b = r / CINP;
  float v = (cin < CIN) ? c[((size_t)b * CIN + cin) * T + t] : 0.f;
  c16[idx] = (_Float16)v;
}

__global__ void pack_block_kernel(
    const float* __restrict__ fw,  const float* __restrict__ fb,
    const float* __restrict__ gw,  const float* __restrict__ gb,
    const float* __restrict__ fcw, const float* __restrict__ fcb,
    const float* __restrict__ gcw, const float* __restrict__ gcb,
    const float* __restrict__ rw,  const float* __restrict__ rb,
    const float* __restrict__ sw,  const float* __restrict__ sb,
    _Float16* __restrict__ Afg, _Float16* __restrict__ Ars,
    float* __restrict__ bfg, float* __restrict__ brs) {
  using namespace cfg;
  const int NA = 512 * KFG;        // gate weights
  const int NR = 256 * 256;        // res/skip weights
  int idx = blockIdx.x * 256 + threadIdx.x;
  if (idx < NA) {
    int m = idx / KFG, k = idx % KFG;
    const float* W  = (m < 256) ? fw  : gw;
    const float* Wc = (m < 256) ? fcw : gcw;
    int mm = m & 255;
    float v;
    if (k < 384) {                       // dilated taps: k = tap*128 + cin
      int tap = k >> 7, cin = k & 127;
      v = W[((size_t)mm * 128 + cin) * 3 + tap];
    } else if (k < 384 + CIN) {          // conditioner 1x1
      v = Wc[(size_t)mm * CIN + (k - 384)];
    } else {
      v = 0.f;                           // pad 80 -> 96
    }
    Afg[idx] = (_Float16)v;
  } else if (idx < NA + NR) {
    int r = idx - NA;
    int m = r >> 8, k = r & 255;
    float v = (m < 128) ? rw[(size_t)m * 256 + k] : sw[(size_t)(m - 128) * 256 + k];
    Ars[r] = (_Float16)v;
  } else if (idx < NA + NR + 512) {
    int m = idx - NA - NR;
    bfg[m] = (m < 256) ? (fb[m] + fcb[m]) : (gb[m - 256] + gcb[m - 256]);
  } else if (idx < NA + NR + 512 + 256) {
    int m = idx - NA - NR - 512;
    brs[m] = (m < 128) ? rb[m] : sb[m - 128];
  }
}

__global__ void pack_f1_kernel(const float* __restrict__ f1w, _Float16* __restrict__ A1) {
  int idx = blockIdx.x * 256 + threadIdx.x;
  if (idx < 128 * 128) A1[idx] = (_Float16)f1w[idx];
}

// ---------------------------------------------------------------------------
// Front causal conv: 1 -> 128 channels, kernel 32, ReLU.  VALU kernel.
// ---------------------------------------------------------------------------
__global__ __launch_bounds__(256) void front_kernel(
    const float* __restrict__ z, const float* __restrict__ fw,
    const float* __restrict__ fb, float* __restrict__ h32,
    _Float16* __restrict__ h16) {
  using namespace cfg;
  int t = blockIdx.x * 256 + threadIdx.x;
  int b = blockIdx.y;
  if (t >= T) return;
  float zv[32];
#pragma unroll
  for (int j = 0; j < 32; ++j) {
    int ts = t - 31 + j;
    zv[j] = (ts >= 0) ? z[(size_t)b * T + ts] : 0.f;
  }
  for (int m = 0; m < RC; ++m) {
    float a = fb[m];
#pragma unroll
    for (int j = 0; j < 32; ++j) a = fmaf(fw[m * 32 + j], zv[j], a);
    a = fmaxf(a, 0.f);
    size_t gi = ((size_t)b * RC + m) * T + t;
    h32[gi] = a;
    h16[gi] = (_Float16)a;
  }
}

// ---------------------------------------------------------------------------
// Gated residual block: both GEMMs + activations fused. 256 thr = 8 waves.
// Tile: 16 time steps, all 512 gate rows / 256 res+skip rows.
// ---------------------------------------------------------------------------
__global__ __launch_bounds__(256) void block_kernel(
    const _Float16* __restrict__ h16in, const float* __restrict__ h32in,
    const _Float16* __restrict__ c16,
    const _Float16* __restrict__ Afg, const float* __restrict__ bfg,
    const _Float16* __restrict__ Ars, const float* __restrict__ brs,
    float* __restrict__ h32out, _Float16* __restrict__ h16out,
    float* __restrict__ skip, int dil) {
  using namespace cfg;
  __shared__ __align__(16) _Float16 bb[16 * 400];     // h taps, [n][k<384], stride 400
  __shared__ __align__(16) _Float16 cb[CINP * 16];    // cond tile, [k'][n] (TDM layout)
  __shared__ __align__(16) _Float16 ob[16 * 264];     // gate output, [n][m], stride 264

  const int t0   = blockIdx.x * 16;
  const int b    = blockIdx.y;
  const int tid  = threadIdx.x;
  const int lane = tid & 31;
  const int w    = tid >> 5;

#if USE_TDM
  // Tensor Data Mover: DMA the 96x16 f16 conditioner tile (row stride T) into LDS.
  if (tid < 32) {
    unsigned long long ga =
        (unsigned long long)(size_t)(c16 + ((size_t)b * CINP) * T + t0);
    unsigned int ldsoff = (unsigned int)(size_t)(void*)cb;
    v4u g0;
    g0[0] = 1u;                                         // count = 1
    g0[1] = ldsoff;                                     // lds_addr
    g0[2] = (unsigned int)(ga & 0xFFFFFFFFu);           // global_addr[31:0]
    g0[3] = (unsigned int)((ga >> 32) & 0x01FFFFFFu)    // global_addr[56:32]
            | 0x80000000u;                              // type = 2 (image)
    v8i g1;
    g1[0] = 0x00010000;                                 // data_size = 1 (2B)
    g1[1] = (int)(((unsigned)T & 0xFFFFu) << 16);       // tensor_dim0 lo16
    g1[2] = (int)(((unsigned)T >> 16) |
                  ((unsigned)CINP << 16));              // dim0 hi16 | tensor_dim1 lo16
    g1[3] = (int)(16u << 16);                           // dim1 hi16 | tile_dim0 = 16
    g1[4] = (int)(unsigned)CINP;                        // tile_dim1 = 96, tile_dim2 = 0
    g1[5] = T;                                          // tensor_dim0_stride lo32
    g1[6] = 0;                                          // stride hi16 | dim1_stride lo16
    g1[7] = 0;                                          // dim1_stride hi (unused, 2D)
    v4i g2 = {0, 0, 0, 0}, g3 = {0, 0, 0, 0};
#if __clang_major__ >= 23
    v8i g4 = {0, 0, 0, 0, 0, 0, 0, 0};
    __builtin_amdgcn_tensor_load_to_lds(g0, g1, g2, g3, g4, 0);
#else
    __builtin_amdgcn_tensor_load_to_lds(g0, g1, g2, g3, 0);
#endif
    __builtin_amdgcn_s_wait_tensorcnt(0);
  }
#else
  for (int idx = tid; idx < CINP * 16; idx += 256) {
    int n = idx & 15, kp = idx >> 4;
    cb[kp * 16 + n] = c16[((size_t)b * CINP + kp) * T + t0 + n];
  }
#endif

  // Stage the 3 dilated causal taps of h: B row k = tap*128+cin holds
  // h[cin][t0+n - dil*(2-tap)], clamped to 0 before t=0.
  for (int idx = tid; idx < 16 * 384; idx += 256) {
    int n = idx & 15, k = idx >> 4;
    int tap = k >> 7, cin = k & 127;
    int t = t0 + n - dil * (2 - tap);
    _Float16 v = (_Float16)0.f;
    if (t >= 0) v = h16in[((size_t)b * RC + cin) * T + t];
    bb[n * 400 + k] = v;
  }
  __syncthreads();

  // ---- GEMM 1: [hf;hg](512 x 16) = Afg(512 x 480) x B(480 x 16) ----
  v8f acc[4];
#pragma unroll
  for (int j = 0; j < 4; ++j) acc[j] = zero8();

  const int mloc = lane & 15;
  for (int kk = 0; kk < KFG / 32; ++kk) {
    const int k0 = kk * 32;
    v16h bf;
    if (kk < 12) {
      bf = load_b_frag(bb, 400, k0, lane);
    } else {                      // conditioner rows live in cb as [k'][n]
      const int hi = lane >> 4, n = lane & 15;
      const int kb = k0 - 384 + hi * 16;
#pragma unroll
      for (int i = 0; i < 16; ++i) bf[i] = cb[(kb + i) * 16 + n];
    }
    if (kk + 1 < KFG / 32)
      __builtin_prefetch(Afg + (size_t)(w * 16 + mloc) * KFG + (kk + 1) * 32, 0, 0);
#pragma unroll
    for (int j = 0; j < 4; ++j) {
      v16h af = load_a_frag(Afg, KFG, (w + j * 8) * 16 + mloc, k0, lane);
      acc[j] = wmma_f16(af, bf, acc[j]);
    }
  }

  // ---- activation: out = tanh(hf) * sigmoid(hg), stash f16 tile in LDS ----
  {
    const int n = lane & 15, hi = lane >> 4;
#pragma unroll
    for (int j = 0; j < 2; ++j) {
#pragma unroll
      for (int r = 0; r < 8; ++r) {
        const int m = j * 128 + w * 16 + hi * 8 + r;   // GC row 0..255
        float fv = acc[j][r]     + bfg[m];
        float gv = acc[j + 2][r] + bfg[256 + m];
        float o  = tanhf(fv) * (1.f / (1.f + __expf(-gv)));
        ob[n * 264 + m] = (_Float16)o;
      }
    }
  }
  __syncthreads();

  // ---- GEMM 2: [res;skip](256 x 16) = Ars(256 x 256) x out(256 x 16) ----
  v8f racc = zero8(), sacc = zero8();
  for (int kk = 0; kk < 8; ++kk) {
    const int k0 = kk * 32;
    v16h bf = load_b_frag(ob, 264, k0, lane);
    v16h ra = load_a_frag(Ars, 256,       w * 16 + mloc, k0, lane);
    v16h sa = load_a_frag(Ars, 256, 128 + w * 16 + mloc, k0, lane);
    racc = wmma_f16(ra, bf, racc);
    sacc = wmma_f16(sa, bf, sacc);
  }

  // ---- epilogue: residual (*sqrt(1/2)) and skip accumulation ----
  {
    const int n = lane & 15, hi = lane >> 4;
    const int t = t0 + n;
#pragma unroll
    for (int r = 0; r < 8; ++r) {
      const int m = w * 16 + hi * 8 + r;               // 0..127
      const size_t gi = ((size_t)b * RC + m) * T + t;
      float hn = (h32in[gi] + racc[r] + brs[m]) * SQRT_HALF;
      h32out[gi] = hn;
      h16out[gi] = (_Float16)hn;
      skip[gi]   = skip[gi] + sacc[r] + brs[128 + m];
    }
  }
}

// ---------------------------------------------------------------------------
// Skip head stage 1: s1 = relu(W1(128x128) * relu(skip) + b1)   (WMMA)
// ---------------------------------------------------------------------------
__global__ __launch_bounds__(256) void final1_kernel(
    const float* __restrict__ skip, const _Float16* __restrict__ A1,
    const float* __restrict__ b1, float* __restrict__ s1) {
  using namespace cfg;
  __shared__ __align__(16) _Float16 sb16[16 * 136];
  const int t0 = blockIdx.x * 16, b = blockIdx.y;
  const int tid = threadIdx.x, lane = tid & 31, w = tid >> 5;
  for (int idx = tid; idx < 16 * 128; idx += 256) {
    int n = idx & 15, k = idx >> 4;
    float v = skip[((size_t)b * SC + k) * T + t0 + n];
    sb16[n * 136 + k] = (_Float16)fmaxf(v, 0.f);
  }
  __syncthreads();
  v8f acc = zero8();
  const int mloc = lane & 15;
  for (int kk = 0; kk < 4; ++kk) {
    v16h bf = load_b_frag(sb16, 136, kk * 32, lane);
    v16h af = load_a_frag(A1, 128, w * 16 + mloc, kk * 32, lane);
    acc = wmma_f16(af, bf, acc);
  }
  const int n = lane & 15, hi = lane >> 4, t = t0 + n;
#pragma unroll
  for (int r = 0; r < 8; ++r) {
    const int m = w * 16 + hi * 8 + r;
    s1[((size_t)b * SC + m) * T + t] = fmaxf(acc[r] + b1[m], 0.f);
  }
}

// Skip head stage 2: ml(2 x T) = W2(2x128) * s1 + b2 (tiny; VALU)
__global__ __launch_bounds__(256) void final2_kernel(
    const float* __restrict__ s1, const float* __restrict__ f2w,
    const float* __restrict__ f2b, float* __restrict__ ml) {
  using namespace cfg;
  int t = blockIdx.x * 256 + threadIdx.x, b = blockIdx.y;
  if (t >= T) return;
  float a0 = f2b[0], a1 = f2b[1];
  for (int k = 0; k < SC; ++k) {
    float v = s1[((size_t)b * SC + k) * T + t];
    a0 = fmaf(f2w[k], v, a0);
    a1 = fmaf(f2w[SC + k], v, a1);
  }
  ml[((size_t)b * 2) * T + t]     = a0;
  ml[((size_t)b * 2 + 1) * T + t] = a1;
}

// IAF elementwise update for one flow.
__global__ __launch_bounds__(256) void iaf_kernel(
    const float* __restrict__ zin, const float* __restrict__ ml,
    float* __restrict__ zout, float* __restrict__ mu_tot,
    float* __restrict__ logs_tot, int first) {
  using namespace cfg;
  int t = blockIdx.x * 256 + threadIdx.x, b = blockIdx.y;
  if (t >= T) return;
  const float* mlb = ml + (size_t)b * 2 * T;
  if (t < T - 1) {
    float mu = mlb[t], lg = mlb[T + t], e = __expf(lg);
    size_t i = (size_t)b * (T - 1) + t;
    if (first) { mu_tot[i] = mu; logs_tot[i] = lg; }
    else       { mu_tot[i] = mu_tot[i] * e + mu; logs_tot[i] += lg; }
  }
  float zv = 0.f;
  if (t > 0) {
    float mu = mlb[t - 1], e = __expf(mlb[T + t - 1]);
    zv = zin[(size_t)b * T + t] * e + mu;   // note: z[:, :, 1:] shifted, padded at t=0
  }
  zout[(size_t)b * T + t] = zv;
}

// ---------------------------------------------------------------------------
// Host orchestration
// ---------------------------------------------------------------------------
extern "C" void kernel_launch(void* const* d_in, const int* in_sizes, int n_in,
                              void* d_out, int out_size, void* d_ws, size_t ws_size,
                              hipStream_t stream) {
  using namespace cfg;
  (void)in_sizes; (void)n_in; (void)out_size; (void)ws_size;

  static const int NBF[4]  = {6, 6, 6, 24};          // blocks per flow
  static const int POW3[6] = {1, 3, 9, 27, 81, 243};

  int cur = 0;
  auto nxt = [&]() { return (const float*)d_in[cur++]; };
  const float* z_in = nxt();
  const float* c_in = nxt();

  struct BlkP { const float *fw,*fb,*gw,*gb,*fcw,*fcb,*gcw,*gcb,*rw,*rb,*sw,*sb; };
  struct FlowP { const float *front_w,*front_b,*f1w,*f1b,*f2w,*f2b; BlkP blk[24]; };
  FlowP fl[4];
  for (int f = 0; f < 4; ++f) {
    fl[f].front_w = nxt(); fl[f].front_b = nxt();
    fl[f].f1w = nxt(); fl[f].f1b = nxt();
    fl[f].f2w = nxt(); fl[f].f2b = nxt();
    for (int i = 0; i < NBF[f]; ++i) {
      BlkP& k = fl[f].blk[i];
      k.fw = nxt();  k.fb = nxt();  k.gw = nxt();  k.gb = nxt();
      k.fcw = nxt(); k.fcb = nxt(); k.gcw = nxt(); k.gcb = nxt();
      k.rw = nxt();  k.rb = nxt();  k.sw = nxt();  k.sb = nxt();
    }
  }

  // -------- workspace carve --------
  char* p = (char*)d_ws;
  auto carve = [&](size_t bytes) {
    void* r = (void*)p;
    p += (bytes + 255) & ~(size_t)255;
    return r;
  };
  const size_t actF = (size_t)B * RC * T;            // 128-channel activation elems
  _Float16* c16    = (_Float16*)carve((size_t)B * CINP * T * 2);
  float*    h32[2] = {(float*)carve(actF * 4), (float*)carve(actF * 4)};
  _Float16* h16[2] = {(_Float16*)carve(actF * 2), (_Float16*)carve(actF * 2)};
  float*    skip   = (float*)carve(actF * 4);
  float*    ml     = (float*)carve((size_t)B * 2 * T * 4);
  float*    zb[2]  = {(float*)carve((size_t)B * T * 4), (float*)carve((size_t)B * T * 4)};
  _Float16 *Afg[42], *Ars[42], *A1[4];
  float    *bfg[42], *brs[42];
  for (int i = 0; i < 42; ++i) {
    Afg[i] = (_Float16*)carve((size_t)512 * KFG * 2);
    Ars[i] = (_Float16*)carve((size_t)256 * 256 * 2);
    bfg[i] = (float*)carve(512 * 4);
    brs[i] = (float*)carve(256 * 4);
  }
  for (int f = 0; f < 4; ++f) A1[f] = (_Float16*)carve((size_t)128 * 128 * 2);
  float* s1 = h32[1];    // free during the skip head (last block output is h32[0])

  // -------- one-time packing (recomputed every call; deterministic) --------
  {
    int tot = B * CINP * T;
    pack_c_kernel<<<dim3((tot + 255) / 256), 256, 0, stream>>>(c_in, c16);
    const int NTOT = 512 * KFG + 256 * 256 + 512 + 256;
    int g = 0;
    for (int f = 0; f < 4; ++f)
      for (int i = 0; i < NBF[f]; ++i, ++g) {
        const BlkP& k = fl[f].blk[i];
        pack_block_kernel<<<dim3((NTOT + 255) / 256), 256, 0, stream>>>(
            k.fw, k.fb, k.gw, k.gb, k.fcw, k.fcb, k.gcw, k.gcb,
            k.rw, k.rb, k.sw, k.sb, Afg[g], Ars[g], bfg[g], brs[g]);
      }
    for (int f = 0; f < 4; ++f)
      pack_f1_kernel<<<dim3(64), 256, 0, stream>>>(fl[f].f1w, A1[f]);
  }

  // -------- output layout: [z | mu_tot | logs_tot] --------
  float* mu_out   = (float*)d_out + (size_t)B * T;
  float* logs_out = mu_out + (size_t)B * (T - 1);

  const dim3 gT16(NT, B), bT(256);
  const dim3 gTall((T + 255) / 256, B);

  const float* zcur = z_in;
  int g = 0;
  for (int f = 0; f < 4; ++f) {
    front_kernel<<<gTall, bT, 0, stream>>>(zcur, fl[f].front_w, fl[f].front_b,
                                           h32[0], h16[0]);
    hipMemsetAsync(skip, 0, actF * 4, stream);
    int cb = 0;
    for (int i = 0; i < NBF[f]; ++i, ++g) {
      block_kernel<<<gT16, bT, 0, stream>>>(h16[cb], h32[cb], c16,
                                            Afg[g], bfg[g], Ars[g], brs[g],
                                            h32[1 - cb], h16[1 - cb], skip,
                                            POW3[i % 6]);
      cb ^= 1;
    }
    final1_kernel<<<gT16, bT, 0, stream>>>(skip, A1[f], fl[f].f1b, s1);
    final2_kernel<<<gTall, bT, 0, stream>>>(s1, fl[f].f2w, fl[f].f2b, ml);
    float* zo = (f == 3) ? (float*)d_out : zb[f & 1];
    iaf_kernel<<<gTall, bT, 0, stream>>>(zcur, ml, zo, mu_out, logs_out, f == 0);
    zcur = zo;
  }
}
